// RNN_74775380623930
// MI455X (gfx1250) — matrix operand order repaired and checked by reading, not compiled, not run on hardware
//
#include <hip/hip_runtime.h>
#include <hip/hip_bf16.h>

typedef _Float16 v16h __attribute__((ext_vector_type(16)));
typedef _Float16 v8h  __attribute__((ext_vector_type(8)));
typedef float    v8f  __attribute__((ext_vector_type(8)));
typedef float    v2f  __attribute__((ext_vector_type(2)));

#define BATCH 1024
#define TLEN  2048
#define HID   32   // hidden size; 4H = 128 gate outputs = 8 N-tiles of 16

#if __has_builtin(__builtin_amdgcn_wmma_f32_16x16x4_f32)
#define HAVE_WMMA_F32X4 1
#else
#define HAVE_WMMA_F32X4 0
#endif

__device__ __forceinline__ float fast_sigmoid(float x) {
    // 1 / (1 + exp2(-x * log2(e)))  -> v_mul + v_exp_f32 + v_add + v_rcp_f32
    return __builtin_amdgcn_rcpf(1.0f + __builtin_amdgcn_exp2f(x * -1.44269504f));
}
__device__ __forceinline__ float fast_tanh(float x) {
    // tanh(x) = 2*sigmoid(2x) - 1
    return __builtin_amdgcn_rcpf(1.0f + __builtin_amdgcn_exp2f(x * -2.88539008f)) * 2.0f - 1.0f;
}

__global__ __launch_bounds__(32)
void lstm_wmma_kernel(const float* __restrict__ x,      // [B, T, 1]
                      const float* __restrict__ W_ih,   // [128, 1]
                      const float* __restrict__ W_hh,   // [128, 32]
                      const float* __restrict__ b_ih,   // [128]
                      const float* __restrict__ b_hh,   // [128]
                      const float* __restrict__ W_out,  // [1, 32]
                      const float* __restrict__ b_out,  // [1]
                      float* __restrict__ out)          // [B, T, 1]
{
    // One wave (32 lanes) per block; each wave owns a 16-row batch tile for all T steps.
    // LDS round-trips within a single wave are in-order (ISA ch.11), so no barriers needed.
    __shared__ _Float16 hbuf[16 * 32];   // h tile staged row-major [row][hcol], 1KB

    const int lane   = threadIdx.x & 31;
    const int lane16 = lane & 15;
    const int halfid = lane >> 4;        // 0: lanes 0-15, 1: lanes 16-31
    const int bbase  = blockIdx.x * 16;

    // ---- Preload B operands: 8 tiles of W_hh^T (32x16 f16 each), resident in 64 VGPRs.
    // B layout (16-bit, 32x16): lane holds column N=lane16; lanes 0-15 K=0..7,16..23, lanes 16-31 K=8..15,24..31.
    v16h WB[8];
#pragma unroll
    for (int n = 0; n < 8; ++n) {
        const float* wr = W_hh + (16 * n + lane16) * HID + 8 * halfid;
#pragma unroll
        for (int e = 0; e < 16; ++e) WB[n][e] = (_Float16)wr[e < 8 ? e : e + 8];
    }

#if HAVE_WMMA_F32X4
    // ---- Input-side tiles for V_WMMA_F32_16X16X4_F32:
    // B (4x16 f32, 2 VGPRs): lanes 0-15 hold K0 (=W_ih[g]) and K1 (=b_ih[g]+b_hh[g]);
    // lanes 16-31 hold K2/K3 (= 0). With A = [x, 1, *, *] the product is exactly the
    // xg = x*W_ih + bias tile in C/D layout, chained into the f16 WMMA's C operand.
    v2f BX[8];
#pragma unroll
    for (int n = 0; n < 8; ++n) {
        const int g = 16 * n + lane16;
        BX[n][0] = (halfid == 0) ? W_ih[g] : 0.0f;
        BX[n][1] = (halfid == 0) ? (b_ih[g] + b_hh[g]) : 0.0f;
    }
#else
    float wih[8], bias[8];
#pragma unroll
    for (int n = 0; n < 8; ++n) {
        const int g = 16 * n + lane16;
        wih[n]  = W_ih[g];
        bias[n] = b_ih[g] + b_hh[g];
    }
#endif

    // Output projection coefficients at this lane's A-layout K positions.
    float wout[16];
#pragma unroll
    for (int e = 0; e < 8; ++e) {
        wout[e]     = W_out[8 * halfid + e];
        wout[8 + e] = W_out[16 + 8 * halfid + e];
    }
    const float bo = b_out[0];

    // ---- State: h (A-layout f16, 8 VGPRs) and c (two v8f in D-layout), start at zero.
    v16h A;
#pragma unroll
    for (int e = 0; e < 16; ++e) A[e] = (_Float16)0;
    v8f c0, c1;
#pragma unroll
    for (int j = 0; j < 8; ++j) { c0[j] = 0.0f; c1[j] = 0.0f; }

    float* orow = out + (size_t)(bbase + lane16) * TLEN;

#if HAVE_WMMA_F32X4
    // A-operand (16x4 f32, 2 VGPRs): lane holds row M=lane16; K0 = x_t[row], K1 = 1.
    // (lanes 16-31 carry K2/K3 which multiply against zeroed B rows.)
    const float* xptr = x + (size_t)(bbase + lane16) * TLEN;
    float xval = xptr[0];
#else
    const float* xrow = x + (size_t)(bbase + 8 * halfid) * TLEN; // rows j+8*halfid, j=0..7
#endif

#pragma unroll 1
    for (int t = 0; t < TLEN; ++t) {
#if HAVE_WMMA_F32X4
        // Software-pipelined prefetch of x_{t+1}, hidden behind this step's work.
        // TLEN is a power of two: wrap to index 0 on the final step (in-bounds, unused).
        const float xnext = xptr[(t + 1) & (TLEN - 1)];
        v2f Axg; Axg[0] = xval; Axg[1] = 1.0f;
        v8f zc;
#pragma unroll
        for (int j = 0; j < 8; ++j) zc[j] = 0.0f;
#else
        float xv[8];
#pragma unroll
        for (int j = 0; j < 8; ++j) xv[j] = xrow[(size_t)j * TLEN + t];
#endif

        // Two h-column groups: group 0 -> hcols 0..15 (tiles 0,2,4,6), group 1 -> hcols 16..31.
#pragma unroll
        for (int grp = 0; grp < 2; ++grp) {
#if HAVE_WMMA_F32X4
            // xg tiles via f32 WMMA (K=4): D = x*W_ih + bias, in C/D layout.
            v8f Ci = __builtin_amdgcn_wmma_f32_16x16x4_f32(false, Axg, false, BX[0 + grp], (short)0, zc, false, false);
            v8f Cf = __builtin_amdgcn_wmma_f32_16x16x4_f32(false, Axg, false, BX[2 + grp], (short)0, zc, false, false);
            v8f Cg = __builtin_amdgcn_wmma_f32_16x16x4_f32(false, Axg, false, BX[4 + grp], (short)0, zc, false, false);
            v8f Co = __builtin_amdgcn_wmma_f32_16x16x4_f32(false, Axg, false, BX[6 + grp], (short)0, zc, false, false);
#else
            v8f Ci, Cf, Cg, Co;
#pragma unroll
            for (int j = 0; j < 8; ++j) {
                Ci[j] = fmaf(xv[j], wih[0 + grp], bias[0 + grp]);
                Cf[j] = fmaf(xv[j], wih[2 + grp], bias[2 + grp]);
                Cg[j] = fmaf(xv[j], wih[4 + grp], bias[4 + grp]);
                Co[j] = fmaf(xv[j], wih[6 + grp], bias[6 + grp]);
            }
#endif
            // gates = h @ W_hh^T + xg   (four chained 16x16x32 f16 WMMAs, in-place accumulate)
            v8f Di = __builtin_amdgcn_wmma_f32_16x16x32_f16(false, A, false, WB[0 + grp], (short)0, Ci, false, false);
            v8f Df = __builtin_amdgcn_wmma_f32_16x16x32_f16(false, A, false, WB[2 + grp], (short)0, Cf, false, false);
            v8f Dg = __builtin_amdgcn_wmma_f32_16x16x32_f16(false, A, false, WB[4 + grp], (short)0, Cg, false, false);
            v8f Do = __builtin_amdgcn_wmma_f32_16x16x32_f16(false, A, false, WB[6 + grp], (short)0, Co, false, false);

            v8f cg = grp ? c1 : c0;
#pragma unroll
            for (int j = 0; j < 8; ++j) {
                const float ii = fast_sigmoid(Di[j]);
                const float ff = fast_sigmoid(Df[j]);
                const float gg = fast_tanh(Dg[j]);
                const float oo = fast_sigmoid(Do[j]);
                const float cc = fmaf(ff, cg[j], ii * gg);
                cg[j] = cc;
                const float hh = oo * fast_tanh(cc);
                // D layout: value at (row = j + 8*halfid, hcol = 16*grp + lane16)
                hbuf[(j + 8 * halfid) * 32 + 16 * grp + lane16] = (_Float16)hh;
            }
            if (grp) c1 = cg; else c0 = cg;
        }

        // Rebuild A operand (16-bit A 16x32 layout) with two ds_load_b128:
        // lanes 0-15: K 0..7 and 16..23 of row lane16; lanes 16-31: K 8..15 and 24..31.
        // Same-wave LDS is processed in order, so the RAW through hbuf needs no barrier;
        // the compiler inserts the s_wait_dscnt before the loaded values are consumed.
        const v8h* lp = (const v8h*)&hbuf[lane16 * 32 + 8 * halfid];
        const v8h lo = lp[0];
        const v8h hi = lp[2]; // +16 halves
        A = __builtin_shufflevector(lo, hi, 0, 1, 2, 3, 4, 5, 6, 7,
                                            8, 9, 10, 11, 12, 13, 14, 15);

        // Output projection: out[b,t] = h . W_out + b_out.
        // Lane l and l+16 hold complementary K-halves of row lane16.
        float s = 0.0f;
#pragma unroll
        for (int e = 0; e < 16; ++e) s = fmaf((float)A[e], wout[e], s);
        s += __shfl_xor(s, 16, 32);
        if (halfid == 0) orow[t] = s + bo;

#if HAVE_WMMA_F32X4
        xval = xnext;
#endif
    }
}

extern "C" void kernel_launch(void* const* d_in, const int* in_sizes, int n_in,
                              void* d_out, int out_size, void* d_ws, size_t ws_size,
                              hipStream_t stream) {
    (void)in_sizes; (void)n_in; (void)out_size; (void)d_ws; (void)ws_size;
    const float* x     = (const float*)d_in[0];
    const float* W_ih  = (const float*)d_in[1];
    const float* W_hh  = (const float*)d_in[2];
    const float* b_ih  = (const float*)d_in[3];
    const float* b_hh  = (const float*)d_in[4];
    const float* W_out = (const float*)d_in[5];
    const float* b_out = (const float*)d_in[6];
    float* out = (float*)d_out;

    dim3 grid(BATCH / 16);   // 64 independent batch-tile chains
    dim3 block(32);          // one wave32 per workgroup
    lstm_wmma_kernel<<<grid, block, 0, stream>>>(x, W_ih, W_hh, b_ih, b_hh, W_out, b_out, out);
}
